// MultiHeadAttention_18485539242200
// MI455X (gfx1250) — matrix operand order repaired and checked by reading, not compiled, not run on hardware
//
#include <hip/hip_runtime.h>

// ---------------------------------------------------------------------------
// MHA forward for MI455X (gfx1250), wave32, WMMA bf16 16x16x32 path.
//   x      [4,2048,1024] f32
//   w_qkv  [3072,1024]   f32   (torch Linear: [out,in], K-contiguous)
//   w_out  [1024,1024]   f32
//   b_out  [1024]        f32
//   out    [4,2048,1024] f32
// Pipeline: gemm_qkv -> flash attention -> gemm_out.  All matmuls on
// v_wmma_f32_16x16x32_bf16 with f32 accumulation.
// ---------------------------------------------------------------------------

typedef __attribute__((ext_vector_type(16))) __bf16 v16bf;
typedef __attribute__((ext_vector_type(8)))  float  v8f;

#define HID   1024
#define NH    16
#define HD    64
#define BT    8192          // B*T rows
#define TSEQ  2048
#define SCALE 0.125f        // 1/sqrt(64)
#define LDSW  56            // LDS tile row stride (elems): 112B, 16B aligned,
                            // optimal 2-cycle bank pattern for per-lane b128

// Native converts: clang lowers __bf16 casts to v_cvt_pk_bf16_f32 on gfx1250.
__device__ __forceinline__ unsigned pack2bf(float a, float b) {
  union { __bf16 h[2]; unsigned u; } u;
  u.h[0] = (__bf16)a; u.h[1] = (__bf16)b;
  return u.u;
}
__device__ __forceinline__ unsigned short f2bf(float f) {
  union { __bf16 h; unsigned short u; } u;
  u.h = (__bf16)f;
  return u.u;
}

__device__ __forceinline__ v8f wmma_bf16(v16bf a, v16bf b, v8f c) {
  return __builtin_amdgcn_wmma_f32_16x16x32_bf16(
      false, a, false, b, (short)0, c, false, false);
}

// A-fragment (16x32 MxK): lane half=l>>4, m=l&15 holds K = (e&8)*2+(e&7)+half*8
// of row m  ==> two aligned 16B loads from a row-major K-contiguous tile.
__device__ __forceinline__ v16bf ldA(const unsigned short* rowbase, int stride,
                                     int k0, int lane) {
  const int hf = (lane >> 4) & 1, m = lane & 15;
  const unsigned short* p = rowbase + m * stride + k0;
  union { uint4 q[2]; v16bf v; } u;
  u.q[0] = *(const uint4*)(p + hf * 8);
  u.q[1] = *(const uint4*)(p + 16 + hf * 8);
  return u.v;
}

// B-fragment (32x16 KxN): lane holds column n=l&15, K = (e&8)*2+(e&7)+half*8
// ==> two aligned 16B loads from a tile stored [N][K] (K-contiguous rows).
__device__ __forceinline__ v16bf ldB(const unsigned short* rowbase, int stride,
                                     int k0, int lane) {
  const int hf = (lane >> 4) & 1, n = lane & 15;
  const unsigned short* p = rowbase + n * stride + k0;
  union { uint4 q[2]; v16bf v; } u;
  u.q[0] = *(const uint4*)(p + hf * 8);
  u.q[1] = *(const uint4*)(p + 16 + hf * 8);
  return u.v;
}

// ---------------------------------------------------------------------------
// Kernel 1: qkv = x @ w_qkv^T, scattered as bf16 into Q[BH,T,D], K[BH,T,D],
// V^T[BH,D,T].  Block 64x64, 4 waves, wave tile 32x32, K-step 32.
// ---------------------------------------------------------------------------
__global__ __launch_bounds__(128) void gemm_qkv(
    const float* __restrict__ x, const float* __restrict__ w,
    unsigned short* __restrict__ qb, unsigned short* __restrict__ kb,
    unsigned short* __restrict__ vtb) {
  __shared__ unsigned short la[64 * LDSW];
  __shared__ unsigned short lb[64 * LDSW];
  const int tid = threadIdx.x, lane = tid & 31, wv = tid >> 5;
  const int m0 = blockIdx.y * 64, n0 = blockIdx.x * 64;
  const int wm = (wv & 1) * 32, wn = (wv >> 1) * 32;

  v8f acc[2][2] = {};

  for (int k0 = 0; k0 < HID; k0 += 32) {
    // stage A(x) and B(w_qkv) 64x32 tiles, converting f32 -> bf16
#pragma unroll
    for (int i = 0; i < 4; ++i) {
      const int f = tid + i * 128;            // 512 float4 chunks
      const int row = f >> 3, c4 = (f & 7) * 4;
      float4 av = *(const float4*)(x + (size_t)(m0 + row) * HID + k0 + c4);
      float4 bv = *(const float4*)(w + (size_t)(n0 + row) * HID + k0 + c4);
      uint2 ap = { pack2bf(av.x, av.y), pack2bf(av.z, av.w) };
      uint2 bp = { pack2bf(bv.x, bv.y), pack2bf(bv.z, bv.w) };
      *(uint2*)(la + row * LDSW + c4) = ap;
      *(uint2*)(lb + row * LDSW + c4) = bp;
    }
    __syncthreads();
    v16bf b0 = ldB(lb + (wn)      * LDSW, LDSW, 0, lane);
    v16bf b1 = ldB(lb + (wn + 16) * LDSW, LDSW, 0, lane);
#pragma unroll
    for (int mf = 0; mf < 2; ++mf) {
      v16bf a = ldA(la + (wm + mf * 16) * LDSW, LDSW, 0, lane);
      acc[mf][0] = wmma_bf16(a, b0, acc[mf][0]);
      acc[mf][1] = wmma_bf16(a, b1, acc[mf][1]);
    }
    __syncthreads();
  }

  // Epilogue: scatter to q / k / v^T (bf16).  All selector math is uniform:
  // the 64-wide N block never straddles a 1024 boundary, and each 16-col
  // fragment never straddles a 64 (head) boundary.
  const int hf = lane >> 4, nn = lane & 15;
  const int which = n0 >> 10;                       // uniform: 0=q, 1=k, 2=v
  const int bb = m0 >> 11, t0 = (m0 & 2047) + wm;   // uniform
  if (which < 2) {
    unsigned short* dst = which ? kb : qb;
#pragma unroll
    for (int nf = 0; nf < 2; ++nf) {
      const int ncol = (n0 & 1023) + wn + nf * 16;  // uniform frag col base
      const size_t bh = (size_t)(bb * NH + (ncol >> 6));
      const int d = (ncol & 63) + nn;
#pragma unroll
      for (int mf = 0; mf < 2; ++mf)
#pragma unroll
        for (int r = 0; r < 8; ++r) {
          const int t = t0 + mf * 16 + hf * 8 + r;
          dst[(bh * TSEQ + t) * HD + d] = f2bf(acc[mf][nf][r]);
        }
    }
  } else {
#pragma unroll
    for (int nf = 0; nf < 2; ++nf) {
      const int ncol = (n0 & 1023) + wn + nf * 16;
      const size_t bh = (size_t)(bb * NH + (ncol >> 6));
      const int d = (ncol & 63) + nn;
#pragma unroll
      for (int mf = 0; mf < 2; ++mf)
#pragma unroll
        for (int r = 0; r < 8; ++r) {
          const int t = t0 + mf * 16 + hf * 8 + r;
          vtb[(bh * HD + d) * TSEQ + t] = f2bf(acc[mf][nf][r]);
        }
    }
  }
}

// ---------------------------------------------------------------------------
// Kernel 2: flash attention.  One wave per 16-row query tile; loops over
// 32-key chunks.  S = Q K^T (WMMA), online softmax on C-fragment layout,
// P re-laid out via a private LDS tile, O += P V (WMMA from V^T).
// The 4 waves of a block share bh, so the per-iteration barriers keep them
// reading the same K/V chunk together (L0/L2 temporal locality).
// ---------------------------------------------------------------------------
__global__ __launch_bounds__(128) void attn_kernel(
    const unsigned short* __restrict__ q, const unsigned short* __restrict__ k,
    const unsigned short* __restrict__ vt, unsigned short* __restrict__ ao) {
  __shared__ unsigned short lp[4 * 16 * LDSW];   // per-wave P tile (16x32)
  const int tid = threadIdx.x, lane = tid & 31, wv = tid >> 5;
  const int bh = blockIdx.y;                      // 0..63
  const int t0 = (blockIdx.x * 4 + wv) * 16;      // query tile base row
  const unsigned short* qp = q  + ((size_t)bh * TSEQ + t0) * HD;
  const unsigned short* kp = k  + (size_t)bh * TSEQ * HD;
  const unsigned short* vp = vt + (size_t)bh * HD * TSEQ;
  unsigned short* pt = lp + wv * 16 * LDSW;
  const int hf = lane >> 4, nn = lane & 15;

  const v16bf aq0 = ldA(qp, HD, 0,  lane);        // Q rows, d = 0..31
  const v16bf aq1 = ldA(qp, HD, 32, lane);        // Q rows, d = 32..63

  v8f o[4] = {};
  float mr[8], lr[8];
#pragma unroll
  for (int r = 0; r < 8; ++r) { mr[r] = -__builtin_inff(); lr[r] = 0.f; }

  for (int it = 0; it < TSEQ / 32; ++it) {
    const int tk = it * 32;
    // S tiles: keys [tk, tk+16) and [tk+16, tk+32)
    v8f s0 = {}, s1 = {};
    s0 = wmma_bf16(aq0, ldB(kp + (size_t)(tk)      * HD, HD, 0,  lane), s0);
    s0 = wmma_bf16(aq1, ldB(kp + (size_t)(tk)      * HD, HD, 32, lane), s0);
    s1 = wmma_bf16(aq0, ldB(kp + (size_t)(tk + 16) * HD, HD, 0,  lane), s1);
    s1 = wmma_bf16(aq1, ldB(kp + (size_t)(tk + 16) * HD, HD, 32, lane), s1);

    // online softmax: row stats live per C-frag element r (row = hf*8+r);
    // reduce across the 16 lanes of the half (the N dimension)
    float corr[8];
#pragma unroll
    for (int r = 0; r < 8; ++r) {
      float v0 = s0[r] * SCALE, v1 = s1[r] * SCALE;
      float mx = fmaxf(v0, v1);
      mx = fmaxf(mx, __shfl_xor(mx, 1));
      mx = fmaxf(mx, __shfl_xor(mx, 2));
      mx = fmaxf(mx, __shfl_xor(mx, 4));
      mx = fmaxf(mx, __shfl_xor(mx, 8));
      const float mn = fmaxf(mr[r], mx);
      const float cr = __expf(mr[r] - mn);        // 0 on first chunk
      mr[r] = mn; corr[r] = cr;
      v0 = __expf(v0 - mn); v1 = __expf(v1 - mn);
      s0[r] = v0; s1[r] = v1;
      float ls = v0 + v1;
      ls += __shfl_xor(ls, 1);
      ls += __shfl_xor(ls, 2);
      ls += __shfl_xor(ls, 4);
      ls += __shfl_xor(ls, 8);
      lr[r] = lr[r] * cr + ls;
    }
#pragma unroll
    for (int df = 0; df < 4; ++df)
#pragma unroll
      for (int r = 0; r < 8; ++r) o[df][r] *= corr[r];

    // C-frag -> A-frag relayout of P through LDS (16 rows x 32 keys)
#pragma unroll
    for (int r = 0; r < 8; ++r) {
      pt[(hf * 8 + r) * LDSW + nn]      = f2bf(s0[r]);
      pt[(hf * 8 + r) * LDSW + 16 + nn] = f2bf(s1[r]);
    }
    __syncthreads();
    const v16bf pa = ldA(pt, LDSW, 0, lane);
#pragma unroll
    for (int df = 0; df < 4; ++df) {
      v16bf b = ldB(vp + (size_t)(df * 16) * TSEQ + tk, TSEQ, 0, lane);
      o[df] = wmma_bf16(pa, b, o[df]);
    }
    __syncthreads();
  }

  // Epilogue: out[b][t][h*64+d] = O * (1/l)  (bf16 into attn_out buffer)
  float inv[8];
#pragma unroll
  for (int r = 0; r < 8; ++r) inv[r] = __builtin_amdgcn_rcpf(lr[r]);
  const int b = bh >> 4, h = bh & 15;
#pragma unroll
  for (int df = 0; df < 4; ++df)
#pragma unroll
    for (int r = 0; r < 8; ++r) {
      const int t = t0 + hf * 8 + r;
      const int d = df * 16 + nn;
      ao[((size_t)(b * TSEQ + t)) * HID + h * HD + d] = f2bf(o[df][r] * inv[r]);
    }
}

// ---------------------------------------------------------------------------
// Kernel 3: out = attn_out @ w_out^T + b_out   (A bf16, B converted f32->bf16)
// ---------------------------------------------------------------------------
__global__ __launch_bounds__(128) void gemm_out(
    const unsigned short* __restrict__ ao, const float* __restrict__ w,
    const float* __restrict__ bias, float* __restrict__ out) {
  __shared__ unsigned short la[64 * LDSW];
  __shared__ unsigned short lb[64 * LDSW];
  const int tid = threadIdx.x, lane = tid & 31, wv = tid >> 5;
  const int m0 = blockIdx.y * 64, n0 = blockIdx.x * 64;
  const int wm = (wv & 1) * 32, wn = (wv >> 1) * 32;

  v8f acc[2][2] = {};

  for (int k0 = 0; k0 < HID; k0 += 32) {
#pragma unroll
    for (int i = 0; i < 4; ++i) {
      const int f = tid + i * 128;
      const int row = f >> 3, c4 = (f & 7) * 4;
      uint2 av = *(const uint2*)(ao + (size_t)(m0 + row) * HID + k0 + c4);
      float4 bv = *(const float4*)(w + (size_t)(n0 + row) * HID + k0 + c4);
      uint2 bp = { pack2bf(bv.x, bv.y), pack2bf(bv.z, bv.w) };
      *(uint2*)(la + row * LDSW + c4) = av;
      *(uint2*)(lb + row * LDSW + c4) = bp;
    }
    __syncthreads();
    v16bf b0 = ldB(lb + (wn)      * LDSW, LDSW, 0, lane);
    v16bf b1 = ldB(lb + (wn + 16) * LDSW, LDSW, 0, lane);
#pragma unroll
    for (int mf = 0; mf < 2; ++mf) {
      v16bf a = ldA(la + (wm + mf * 16) * LDSW, LDSW, 0, lane);
      acc[mf][0] = wmma_bf16(a, b0, acc[mf][0]);
      acc[mf][1] = wmma_bf16(a, b1, acc[mf][1]);
    }
    __syncthreads();
  }

  const int hf = lane >> 4, nn = lane & 15;
#pragma unroll
  for (int mf = 0; mf < 2; ++mf)
#pragma unroll
    for (int nf = 0; nf < 2; ++nf)
#pragma unroll
      for (int r = 0; r < 8; ++r) {
        const int m = m0 + wm + mf * 16 + hf * 8 + r;
        const int n = n0 + wn + nf * 16 + nn;
        out[(size_t)m * HID + n] = acc[mf][nf][r] + bias[n];
      }
}

// ---------------------------------------------------------------------------
extern "C" void kernel_launch(void* const* d_in, const int* in_sizes, int n_in,
                              void* d_out, int out_size, void* d_ws,
                              size_t ws_size, hipStream_t stream) {
  const float* x     = (const float*)d_in[0];
  const float* w_qkv = (const float*)d_in[1];
  const float* w_out = (const float*)d_in[2];
  const float* b_out = (const float*)d_in[3];
  float* out = (float*)d_out;

  // workspace carve (bf16): q, k, v^T, attn_out  = 4 x 16 MB = 64 MB
  const size_t NE = (size_t)BT * HID;   // 8,388,608 elems each
  unsigned short* qb  = (unsigned short*)d_ws;
  unsigned short* kb  = qb + NE;
  unsigned short* vtb = kb + NE;
  unsigned short* aob = vtb + NE;

  gemm_qkv<<<dim3(48, 128), 128, 0, stream>>>(x, w_qkv, qb, kb, vtb);
  attn_kernel<<<dim3(32, 64), 128, 0, stream>>>(qb, kb, vtb, aob);
  gemm_out<<<dim3(16, 128), 128, 0, stream>>>(aob, w_out, b_out, out);
}
